// MixedAttention_85341000172350
// MI455X (gfx1250) — matrix operand order repaired and checked by reading, not compiled, not run on hardware
//
#include <hip/hip_runtime.h>

typedef __bf16 bf16_t;
typedef __attribute__((ext_vector_type(8)))  __bf16 v8bf;
typedef __attribute__((ext_vector_type(16))) __bf16 v16bf;
typedef __attribute__((ext_vector_type(8)))  float  v8f;

constexpr int kB     = 2;
constexpr int kC     = 256;
constexpr int kH     = 56;
constexpr int kW     = 56;
constexpr int kHW    = kH * kW;          // 3136
constexpr int kM     = kB * kHW;         // 6272 rows in [pos][chan] matrices
constexpr int kHeads = 8;
constexpr int kDk    = 32;
constexpr float kEps   = 1e-5f;
constexpr float kSlope = 0.01f;

// ---------------- WMMA helpers ----------------

__device__ __forceinline__ v8f wmma_bf16(v16bf a, v16bf b, v8f c) {
  return __builtin_amdgcn_wmma_f32_16x16x32_bf16(false, a, false, b, (short)0, c,
                                                 false, false);
}

__device__ __forceinline__ v16bf join8(v8bf lo, v8bf hi) {
  return __builtin_shufflevector(lo, hi, 0, 1, 2, 3, 4, 5, 6, 7,
                                 8, 9, 10, 11, 12, 13, 14, 15);
}

__device__ __forceinline__ v8f vzero8() {
  v8f z;
#pragma unroll
  for (int i = 0; i < 8; ++i) z[i] = 0.0f;
  return z;
}

// A-matrix fragment (16x32 bf16): lane = row (l&15), K elems {kb8..+7, kb8+16..+23}
__device__ __forceinline__ v16bf load_a(const bf16_t* A, int lda, int row0, int k0) {
  const int lane = threadIdx.x & 31;
  const bf16_t* p = A + (size_t)(row0 + (lane & 15)) * lda + (k0 + ((lane >> 4) << 3));
  return join8(*(const v8bf*)p, *(const v8bf*)(p + 16));
}

// A fragment with per-lane absolute source row (srcRow < 0 -> zeros), for conv taps
__device__ __forceinline__ v16bf load_a_guard(const bf16_t* A, int lda, int srcRow, int k0) {
  const int lane = threadIdx.x & 31;
  v16bf f;
#pragma unroll
  for (int j = 0; j < 16; ++j) f[j] = (bf16_t)0.0f;
  if (srcRow >= 0) {
    const bf16_t* p = A + (size_t)srcRow * lda + (k0 + ((lane >> 4) << 3));
    f = join8(*(const v8bf*)p, *(const v8bf*)(p + 16));
  }
  return f;
}

// B-matrix fragment from row-major W[n][k] (contraction contiguous):
// lane = column (l&15), contraction elems {kb8..+7, kb8+16..+23}
__device__ __forceinline__ v16bf load_b(const bf16_t* Wm, int ldw, int col0, int k0) {
  const int lane = threadIdx.x & 31;
  const bf16_t* p = Wm + (size_t)(col0 + (lane & 15)) * ldw + (k0 + ((lane >> 4) << 3));
  return join8(*(const v8bf*)p, *(const v8bf*)(p + 16));
}

// ---------------- packing kernels ----------------

__global__ __launch_bounds__(256) void k_pack_x(const float* __restrict__ x,
                                                bf16_t* __restrict__ Xp) {
  const int idx = blockIdx.x * 256 + threadIdx.x;   // over kM*kC
  const int c = idx & (kC - 1);
  const int R = idx >> 8;
  const int b = R / kHW, p = R - b * kHW;
  Xp[idx] = (bf16_t)x[((size_t)b * kC + c) * kHW + p];
}

__global__ __launch_bounds__(256) void k_f2b(const float* __restrict__ s,
                                             bf16_t* __restrict__ d, int n) {
  const int i = blockIdx.x * 256 + threadIdx.x;
  if (i < n) d[i] = (bf16_t)s[i];
}

// ksw [O,C,3,3] f32  ->  w9 [tap][O][C] bf16 (contraction-contiguous per tap)
__global__ __launch_bounds__(256) void k_pack_ks(const float* __restrict__ ksw,
                                                 bf16_t* __restrict__ w9) {
  const int i = blockIdx.x * 256 + threadIdx.x;     // over 9*kC*kC
  const int t = i / (kC * kC);
  const int rem = i - t * kC * kC;
  const int o = rem >> 8, c = rem & (kC - 1);
  w9[i] = (bf16_t)ksw[((size_t)(o * kC + c) * 3 + (t / 3)) * 3 + (t % 3)];
}

// ---------------- fused Q/K/V GEMM ----------------
// grid (kM/64, kC/128), block 256 (8 waves: wave&1 -> M subtile, wave>>1 -> N subtile)
// Also emits Vt[b][head][d][pos] so attention's P*V B-fragments are contiguous.

__global__ __launch_bounds__(256) void k_qkv(
    const bf16_t* __restrict__ Xp,
    const bf16_t* __restrict__ wq, const bf16_t* __restrict__ wk,
    const bf16_t* __restrict__ wv,
    const float* __restrict__ qbias, const float* __restrict__ kbias,
    const float* __restrict__ vbias,
    bf16_t* __restrict__ Qp, bf16_t* __restrict__ Kp, bf16_t* __restrict__ Vp,
    bf16_t* __restrict__ Vt) {
  const int wave = threadIdx.x >> 5, lane = threadIdx.x & 31;
  const int row0 = blockIdx.x * 64 + (wave & 1) * 32;
  const int col0 = blockIdx.y * 128 + (wave >> 1) * 32;
  v8f acc[3][2][2];
#pragma unroll
  for (int m = 0; m < 3; ++m)
#pragma unroll
    for (int i = 0; i < 2; ++i)
#pragma unroll
      for (int j = 0; j < 2; ++j) acc[m][i][j] = vzero8();

  for (int k0 = 0; k0 < kC; k0 += 32) {
    const v16bf a0 = load_a(Xp, kC, row0, k0);
    const v16bf a1 = load_a(Xp, kC, row0 + 16, k0);
    v16bf bfr;
    bfr = load_b(wq, kC, col0, k0);
    acc[0][0][0] = wmma_bf16(a0, bfr, acc[0][0][0]);
    acc[0][1][0] = wmma_bf16(a1, bfr, acc[0][1][0]);
    bfr = load_b(wq, kC, col0 + 16, k0);
    acc[0][0][1] = wmma_bf16(a0, bfr, acc[0][0][1]);
    acc[0][1][1] = wmma_bf16(a1, bfr, acc[0][1][1]);
    bfr = load_b(wk, kC, col0, k0);
    acc[1][0][0] = wmma_bf16(a0, bfr, acc[1][0][0]);
    acc[1][1][0] = wmma_bf16(a1, bfr, acc[1][1][0]);
    bfr = load_b(wk, kC, col0 + 16, k0);
    acc[1][0][1] = wmma_bf16(a0, bfr, acc[1][0][1]);
    acc[1][1][1] = wmma_bf16(a1, bfr, acc[1][1][1]);
    bfr = load_b(wv, kC, col0, k0);
    acc[2][0][0] = wmma_bf16(a0, bfr, acc[2][0][0]);
    acc[2][1][0] = wmma_bf16(a1, bfr, acc[2][1][0]);
    bfr = load_b(wv, kC, col0 + 16, k0);
    acc[2][0][1] = wmma_bf16(a0, bfr, acc[2][0][1]);
    acc[2][1][1] = wmma_bf16(a1, bfr, acc[2][1][1]);
  }

#pragma unroll
  for (int i = 0; i < 2; ++i) {
    const int rowb = row0 + i * 16 + ((lane >> 4) << 3);
    const int bat = rowb / kHW;              // 8-row span never crosses batch
    const int pbase = rowb - bat * kHW;
#pragma unroll
    for (int j = 0; j < 2; ++j) {
      const int col = col0 + j * 16 + (lane & 15);
      const float bq = qbias[col], bk = kbias[col], bv = vbias[col];
      const int hd = col >> 5, dd = col & (kDk - 1);
      bf16_t* vtrow = Vt + ((size_t)(bat * kHeads + hd) * kDk + dd) * kHW + pbase;
#pragma unroll
      for (int r = 0; r < 8; ++r) {
        const size_t o = (size_t)(rowb + r) * kC + col;
        const float vvl = acc[2][i][j][r] + bv;
        Qp[o] = (bf16_t)(acc[0][i][j][r] + bq);
        Kp[o] = (bf16_t)(acc[1][i][j][r] + bk);
        Vp[o] = (bf16_t)vvl;
        vtrow[r] = (bf16_t)vvl;
      }
    }
  }
}

// ---------------- 3x3 conv as 9 shifted GEMMs ----------------

__global__ __launch_bounds__(256) void k_conv3(
    const bf16_t* __restrict__ Xp, const bf16_t* __restrict__ wks9,
    const float* __restrict__ ksb, bf16_t* __restrict__ Ksp) {
  const int wave = threadIdx.x >> 5, lane = threadIdx.x & 31;
  const int row0 = blockIdx.x * 64 + (wave & 1) * 32;
  const int col0 = blockIdx.y * 128 + (wave >> 1) * 32;
  v8f acc[2][2];
#pragma unroll
  for (int i = 0; i < 2; ++i)
#pragma unroll
    for (int j = 0; j < 2; ++j) acc[i][j] = vzero8();

  const int R0 = row0 + (lane & 15);
  const int R1 = R0 + 16;
  const int bA = R0 / kHW, pA = R0 - bA * kHW, yA = pA / kW, xA = pA - yA * kW;
  const int bBt = R1 / kHW, pBt = R1 - bBt * kHW, yB = pBt / kW, xB = pBt - yB * kW;

  for (int t = 0; t < 9; ++t) {
    const int dy = t / 3 - 1, dx = t % 3 - 1;
    const int ya = yA + dy, xa = xA + dx;
    const int yb = yB + dy, xb = xB + dx;
    const int sr0 = (ya >= 0 && ya < kH && xa >= 0 && xa < kW)
                        ? (bA * kHW + ya * kW + xa) : -1;
    const int sr1 = (yb >= 0 && yb < kH && xb >= 0 && xb < kW)
                        ? (bBt * kHW + yb * kW + xb) : -1;
    const bf16_t* wt = wks9 + (size_t)t * kC * kC;
    for (int k0 = 0; k0 < kC; k0 += 32) {
      const v16bf a0 = load_a_guard(Xp, kC, sr0, k0);
      const v16bf a1 = load_a_guard(Xp, kC, sr1, k0);
      v16bf bfr = load_b(wt, kC, col0, k0);
      acc[0][0] = wmma_bf16(a0, bfr, acc[0][0]);
      acc[1][0] = wmma_bf16(a1, bfr, acc[1][0]);
      bfr = load_b(wt, kC, col0 + 16, k0);
      acc[0][1] = wmma_bf16(a0, bfr, acc[0][1]);
      acc[1][1] = wmma_bf16(a1, bfr, acc[1][1]);
    }
  }

#pragma unroll
  for (int i = 0; i < 2; ++i) {
    const int rowb = row0 + i * 16 + ((lane >> 4) << 3);
#pragma unroll
    for (int j = 0; j < 2; ++j) {
      const int col = col0 + j * 16 + (lane & 15);
      const float bias = ksb[col];
#pragma unroll
      for (int r = 0; r < 8; ++r)
        Ksp[(size_t)(rowb + r) * kC + col] = (bf16_t)(acc[i][j][r] + bias);
    }
  }
}

// ---------------- flash attention: sa half of output ----------------
// grid (49, kB*kHeads), block 128 (4 waves, one 16-query tile per wave)

__global__ __launch_bounds__(128) void k_attn(
    const bf16_t* __restrict__ Qp, const bf16_t* __restrict__ Kp,
    const bf16_t* __restrict__ Vt, float* __restrict__ out) {
  __shared__ bf16_t plds[4][16][40];   // P tile transpose staging, 16B-aligned rows
  const int wave = threadIdx.x >> 5, lane = threadIdx.x & 31;
  const int bat = blockIdx.y >> 3, hd = blockIdx.y & 7;
  const int q0 = (blockIdx.x * 4 + wave) * 16;
  const int rowQ = bat * kHW + q0;
  const int cb = hd * kDk;
  const bf16_t* Vth = Vt + (size_t)(bat * kHeads + hd) * kDk * kHW; // [d][pos]
  const float scale = 0.1767766952966369f;  // 1/sqrt(32)

  const v16bf aq = load_a(Qp, kC, rowQ, cb);
  v8f o0 = vzero8(), o1 = vzero8();
  float mrow[8], lrow[8];
#pragma unroll
  for (int r = 0; r < 8; ++r) { mrow[r] = -3.0e38f; lrow[r] = 0.0f; }

  for (int m0 = 0; m0 < kHW; m0 += 32) {
    const int rowK = bat * kHW + m0;
    v8f s0 = vzero8(), s1 = vzero8();
    s0 = wmma_bf16(aq, load_b(Kp, kC, rowK, cb), s0);
    s1 = wmma_bf16(aq, load_b(Kp, kC, rowK + 16, cb), s1);

    const int colm = lane & 15;
    const int rhi = (lane >> 4) << 3;
#pragma unroll
    for (int r = 0; r < 8; ++r) {
      const float v0 = s0[r] * scale;
      const float v1 = s1[r] * scale;
      float rm = fmaxf(v0, v1);
      rm = fmaxf(rm, __shfl_xor(rm, 1));
      rm = fmaxf(rm, __shfl_xor(rm, 2));
      rm = fmaxf(rm, __shfl_xor(rm, 4));
      rm = fmaxf(rm, __shfl_xor(rm, 8));
      const float nm = fmaxf(mrow[r], rm);
      const float alpha = __expf(mrow[r] - nm);
      mrow[r] = nm;
      const float p0 = __expf(v0 - nm);
      const float p1 = __expf(v1 - nm);
      plds[wave][rhi + r][colm] = (bf16_t)p0;
      plds[wave][rhi + r][colm + 16] = (bf16_t)p1;
      float rs = p0 + p1;
      rs += __shfl_xor(rs, 1);
      rs += __shfl_xor(rs, 2);
      rs += __shfl_xor(rs, 4);
      rs += __shfl_xor(rs, 8);
      lrow[r] = lrow[r] * alpha + rs;
      o0[r] *= alpha;
      o1[r] *= alpha;
    }
    __syncthreads();

    v16bf ap;
    {
      const int row = lane & 15, kb8 = (lane >> 4) << 3;
      const bf16_t* pr = &plds[wave][row][0];
#pragma unroll
      for (int j = 0; j < 8; ++j) { ap[j] = pr[kb8 + j]; ap[8 + j] = pr[kb8 + 16 + j]; }
    }
    // P*V: B-fragment from Vt[d][pos] -> contraction over pos is contiguous
    o0 = wmma_bf16(ap, load_b(Vth, kHW, 0, m0), o0);
    o1 = wmma_bf16(ap, load_b(Vth, kHW, 16, m0), o1);
    __syncthreads();
  }

  const int d = lane & 15;
  const int rhi = (lane >> 4) << 3;
#pragma unroll
  for (int r = 0; r < 8; ++r) {
    const int q = q0 + rhi + r;
    const float inv = 1.0f / lrow[r];
    out[((size_t)bat * (2 * kC) + cb + d) * kHW + q] = o0[r] * inv;
    out[((size_t)bat * (2 * kC) + cb + 16 + d) * kHW + q] = o1[r] * inv;
  }
}

// ---------------- fused gate: vs = V * sigmoid(sd1(Q.*Ks)+b) ----------------

__global__ __launch_bounds__(256) void k_gate_sd1(
    const bf16_t* __restrict__ Qp, const bf16_t* __restrict__ Ksp,
    const bf16_t* __restrict__ Vp, const bf16_t* __restrict__ wsd1,
    const float* __restrict__ sd1b, bf16_t* __restrict__ VSp) {
  const int wave = threadIdx.x >> 5, lane = threadIdx.x & 31;
  const int row0 = blockIdx.x * 64 + (wave & 1) * 32;
  const int col0 = blockIdx.y * 128 + (wave >> 1) * 32;
  v8f acc[2][2];
#pragma unroll
  for (int i = 0; i < 2; ++i)
#pragma unroll
    for (int j = 0; j < 2; ++j) acc[i][j] = vzero8();

  for (int k0 = 0; k0 < kC; k0 += 32) {
    v16bf aq0 = load_a(Qp, kC, row0, k0);
    v16bf as0 = load_a(Ksp, kC, row0, k0);
    v16bf aq1 = load_a(Qp, kC, row0 + 16, k0);
    v16bf as1 = load_a(Ksp, kC, row0 + 16, k0);
    v16bf a0, a1;
#pragma unroll
    for (int j = 0; j < 16; ++j) {
      a0[j] = (bf16_t)((float)aq0[j] * (float)as0[j]);
      a1[j] = (bf16_t)((float)aq1[j] * (float)as1[j]);
    }
    v16bf bfr = load_b(wsd1, kC, col0, k0);
    acc[0][0] = wmma_bf16(a0, bfr, acc[0][0]);
    acc[1][0] = wmma_bf16(a1, bfr, acc[1][0]);
    bfr = load_b(wsd1, kC, col0 + 16, k0);
    acc[0][1] = wmma_bf16(a0, bfr, acc[0][1]);
    acc[1][1] = wmma_bf16(a1, bfr, acc[1][1]);
  }

#pragma unroll
  for (int i = 0; i < 2; ++i) {
    const int rowb = row0 + i * 16 + ((lane >> 4) << 3);
#pragma unroll
    for (int j = 0; j < 2; ++j) {
      const int col = col0 + j * 16 + (lane & 15);
      const float bias = sd1b[col];
#pragma unroll
      for (int r = 0; r < 8; ++r) {
        const size_t o = (size_t)(rowb + r) * kC + col;
        const float g = 1.0f / (1.0f + __expf(-(acc[i][j][r] + bias)));
        VSp[o] = (bf16_t)((float)Vp[o] * g);
      }
    }
  }
}

// ---------------- depthwise 3x3 + BN1 + LeakyReLU ----------------

__global__ __launch_bounds__(256) void k_dw_bn1(
    const bf16_t* __restrict__ VSp, const float* __restrict__ dww,
    const float* __restrict__ g1, const float* __restrict__ b1,
    const float* __restrict__ m1, const float* __restrict__ v1,
    bf16_t* __restrict__ Y1) {
  const int idx = blockIdx.x * 256 + threadIdx.x;  // over kM*kC
  const int c = idx & (kC - 1);
  const int R = idx >> 8;
  const int b = R / kHW, p = R - b * kHW, y = p / kW, x = p - y * kW;
  float s = 0.0f;
#pragma unroll
  for (int dy = -1; dy <= 1; ++dy)
#pragma unroll
    for (int dx = -1; dx <= 1; ++dx) {
      const int yy = y + dy, xx = x + dx;
      if (yy < 0 || yy >= kH || xx < 0 || xx >= kW) continue;
      s += (float)VSp[(size_t)(b * kHW + yy * kW + xx) * kC + c] *
           dww[c * 9 + (dy + 1) * 3 + (dx + 1)];
    }
  const float sc = g1[c] * rsqrtf(v1[c] + kEps);
  const float t = (s - m1[c]) * sc + b1[c];
  Y1[(size_t)R * kC + c] = (bf16_t)(t > 0.0f ? t : kSlope * t);
}

// ---------------- pointwise GEMM + BN2 + LeakyReLU ----------------

__global__ __launch_bounds__(256) void k_pw_bn2(
    const bf16_t* __restrict__ Y1, const bf16_t* __restrict__ wpw,
    const float* __restrict__ g2, const float* __restrict__ b2,
    const float* __restrict__ m2, const float* __restrict__ v2,
    bf16_t* __restrict__ Y2) {
  const int wave = threadIdx.x >> 5, lane = threadIdx.x & 31;
  const int row0 = blockIdx.x * 64 + (wave & 1) * 32;
  const int col0 = blockIdx.y * 128 + (wave >> 1) * 32;
  v8f acc[2][2];
#pragma unroll
  for (int i = 0; i < 2; ++i)
#pragma unroll
    for (int j = 0; j < 2; ++j) acc[i][j] = vzero8();

  for (int k0 = 0; k0 < kC; k0 += 32) {
    const v16bf a0 = load_a(Y1, kC, row0, k0);
    const v16bf a1 = load_a(Y1, kC, row0 + 16, k0);
    v16bf bfr = load_b(wpw, kC, col0, k0);
    acc[0][0] = wmma_bf16(a0, bfr, acc[0][0]);
    acc[1][0] = wmma_bf16(a1, bfr, acc[1][0]);
    bfr = load_b(wpw, kC, col0 + 16, k0);
    acc[0][1] = wmma_bf16(a0, bfr, acc[0][1]);
    acc[1][1] = wmma_bf16(a1, bfr, acc[1][1]);
  }

#pragma unroll
  for (int i = 0; i < 2; ++i) {
    const int rowb = row0 + i * 16 + ((lane >> 4) << 3);
#pragma unroll
    for (int j = 0; j < 2; ++j) {
      const int col = col0 + j * 16 + (lane & 15);
      const float sc = g2[col] * rsqrtf(v2[col] + kEps);
      const float mm = m2[col], bb = b2[col];
#pragma unroll
      for (int r = 0; r < 8; ++r) {
        const float t = (acc[i][j][r] - mm) * sc + bb;
        Y2[(size_t)(rowb + r) * kC + col] = (bf16_t)(t > 0.0f ? t : kSlope * t);
      }
    }
  }
}

// ---------------- final sd2 GEMM -> sd half of output (NCHW f32) ----------------

__global__ __launch_bounds__(256) void k_sd2(
    const bf16_t* __restrict__ Y2, const bf16_t* __restrict__ wsd2,
    const float* __restrict__ sd2b, float* __restrict__ out) {
  const int wave = threadIdx.x >> 5, lane = threadIdx.x & 31;
  const int row0 = blockIdx.x * 64 + (wave & 1) * 32;
  const int col0 = blockIdx.y * 128 + (wave >> 1) * 32;
  v8f acc[2][2];
#pragma unroll
  for (int i = 0; i < 2; ++i)
#pragma unroll
    for (int j = 0; j < 2; ++j) acc[i][j] = vzero8();

  for (int k0 = 0; k0 < kC; k0 += 32) {
    const v16bf a0 = load_a(Y2, kC, row0, k0);
    const v16bf a1 = load_a(Y2, kC, row0 + 16, k0);
    v16bf bfr = load_b(wsd2, kC, col0, k0);
    acc[0][0] = wmma_bf16(a0, bfr, acc[0][0]);
    acc[1][0] = wmma_bf16(a1, bfr, acc[1][0]);
    bfr = load_b(wsd2, kC, col0 + 16, k0);
    acc[0][1] = wmma_bf16(a0, bfr, acc[0][1]);
    acc[1][1] = wmma_bf16(a1, bfr, acc[1][1]);
  }

#pragma unroll
  for (int i = 0; i < 2; ++i) {
    const int rowb = row0 + i * 16 + ((lane >> 4) << 3);
    const int b = rowb / kHW, pbase = rowb - b * kHW;
#pragma unroll
    for (int j = 0; j < 2; ++j) {
      const int col = col0 + j * 16 + (lane & 15);
      const float bias = sd2b[col];
      float* orow = out + ((size_t)b * (2 * kC) + kC + col) * kHW + pbase;
#pragma unroll
      for (int r = 0; r < 8; ++r) orow[r] = acc[i][j][r] + bias;
    }
  }
}

// ---------------- launch ----------------

extern "C" void kernel_launch(void* const* d_in, const int* in_sizes, int n_in,
                              void* d_out, int out_size, void* d_ws, size_t ws_size,
                              hipStream_t stream) {
  const float* x    = (const float*)d_in[0];
  const float* qw   = (const float*)d_in[1];
  const float* qb   = (const float*)d_in[2];
  const float* kw   = (const float*)d_in[3];
  const float* kb   = (const float*)d_in[4];
  const float* vw   = (const float*)d_in[5];
  const float* vb   = (const float*)d_in[6];
  const float* ksw  = (const float*)d_in[7];
  const float* ksb  = (const float*)d_in[8];
  const float* sd1w = (const float*)d_in[9];
  const float* sd1b = (const float*)d_in[10];
  const float* sd2w = (const float*)d_in[11];
  const float* sd2b = (const float*)d_in[12];
  const float* dww  = (const float*)d_in[13];
  const float* bn1g = (const float*)d_in[14];
  const float* bn1b = (const float*)d_in[15];
  const float* bn1m = (const float*)d_in[16];
  const float* bn1v = (const float*)d_in[17];
  const float* pww  = (const float*)d_in[18];
  const float* bn2g = (const float*)d_in[19];
  const float* bn2b = (const float*)d_in[20];
  const float* bn2m = (const float*)d_in[21];
  const float* bn2v = (const float*)d_in[22];
  float* out = (float*)d_out;

  const size_t ACT = (size_t)kM * kC;             // 1,605,632 elems
  const size_t WSZ = (size_t)kC * kC;             // 65,536 elems
  bf16_t* wsb = (bf16_t*)d_ws;
  bf16_t* Xp   = wsb + 0 * ACT;
  bf16_t* Qp   = wsb + 1 * ACT;
  bf16_t* Kp   = wsb + 2 * ACT;
  bf16_t* Vp   = wsb + 3 * ACT;
  bf16_t* Vt   = wsb + 4 * ACT;                   // [b][head][d][pos]
  bf16_t* Ksp  = wsb + 5 * ACT;
  bf16_t* VSp  = wsb + 6 * ACT;
  bf16_t* Y1   = wsb + 7 * ACT;
  bf16_t* Y2   = wsb + 8 * ACT;
  bf16_t* wqb  = wsb + 9 * ACT;
  bf16_t* wkb  = wqb + WSZ;
  bf16_t* wvb  = wkb + WSZ;
  bf16_t* w1b  = wvb + WSZ;
  bf16_t* w2b  = w1b + WSZ;
  bf16_t* wpb  = w2b + WSZ;
  bf16_t* wk9  = wpb + WSZ;                       // 9*WSZ elems

  const size_t need = (9 * ACT + 15 * WSZ) * sizeof(bf16_t);
  if (ws_size < need) return;

  // pack
  k_pack_x<<<kM * kC / 256, 256, 0, stream>>>(x, Xp);
  k_f2b<<<(int)(WSZ / 256), 256, 0, stream>>>(qw, wqb, (int)WSZ);
  k_f2b<<<(int)(WSZ / 256), 256, 0, stream>>>(kw, wkb, (int)WSZ);
  k_f2b<<<(int)(WSZ / 256), 256, 0, stream>>>(vw, wvb, (int)WSZ);
  k_f2b<<<(int)(WSZ / 256), 256, 0, stream>>>(sd1w, w1b, (int)WSZ);
  k_f2b<<<(int)(WSZ / 256), 256, 0, stream>>>(sd2w, w2b, (int)WSZ);
  k_f2b<<<(int)(WSZ / 256), 256, 0, stream>>>(pww, wpb, (int)WSZ);
  k_pack_ks<<<(int)(9 * WSZ / 256), 256, 0, stream>>>(ksw, wk9);

  const dim3 gGemm(kM / 64, kC / 128);
  // projections
  k_qkv<<<gGemm, 256, 0, stream>>>(Xp, wqb, wkb, wvb, qb, kb, vb, Qp, Kp, Vp, Vt);
  k_conv3<<<gGemm, 256, 0, stream>>>(Xp, wk9, ksb, Ksp);
  // attention (sa half of out)
  k_attn<<<dim3(kHW / 16 / 4, kB * kHeads), 128, 0, stream>>>(Qp, Kp, Vt, out);
  // gated depthwise-separable branch (sd half of out)
  k_gate_sd1<<<gGemm, 256, 0, stream>>>(Qp, Ksp, Vp, w1b, sd1b, VSp);
  k_dw_bn1<<<kM * kC / 256, 256, 0, stream>>>(VSp, dww, bn1g, bn1b, bn1m, bn1v, Y1);
  k_pw_bn2<<<gGemm, 256, 0, stream>>>(Y1, wpb, bn2g, bn2b, bn2m, bn2v, Y2);
  k_sd2<<<gGemm, 256, 0, stream>>>(Y2, w2b, sd2b, out);
}
